// RNN_39084202393635
// MI455X (gfx1250) — matrix operand order repaired and checked by reading, not compiled, not run on hardware
//
#include <hip/hip_runtime.h>
#include <hip/hip_bf16.h>

// ---------------- problem constants ----------------
#define VSZ 50257
#define H   1024
#define OSZ 1024
#define BSZ 64
#define TSZ 256

typedef __bf16 bf16;
typedef __attribute__((ext_vector_type(16))) __bf16 bf16x16;
typedef __attribute__((ext_vector_type(8)))  __bf16 bf16x8;
typedef __attribute__((ext_vector_type(8)))  float  f32x8;

struct alignas(8) bf16x4s { bf16 v[4]; };

__device__ __forceinline__ bf16 f2bf(float f) {
  unsigned u = __builtin_bit_cast(unsigned, f);
  unsigned r = (u + 0x7FFFu + ((u >> 16) & 1u)) >> 16;
  unsigned short s = (unsigned short)r;
  return __builtin_bit_cast(bf16, s);
}

// Build a 16-element bf16 fragment from two contiguous 8-element chunks.
__device__ __forceinline__ bf16x16 load_frag(const bf16* p0, const bf16* p1) {
  bf16x8 lo = *(const bf16x8*)p0;
  bf16x8 hi = *(const bf16x8*)p1;
  bf16x16 r;
#pragma unroll
  for (int i = 0; i < 8; ++i) { r[i] = lo[i]; r[i + 8] = hi[i]; }
  return r;
}

#define WMMA_BF16(a, b, c) \
  __builtin_amdgcn_wmma_f32_16x16x32_bf16(false, (a), false, (b), (short)0, (c), false, false)

// 16x16 output tile GEMM core over K=H with TWO independent accumulator
// chains (halves the WMMA RAW hazard critical path in the serial scan).
__device__ __forceinline__ f32x8 tile_gemm(const bf16* __restrict__ aRow,
                                           const bf16* __restrict__ bRow,
                                           int kA, int kB) {
  f32x8 acc0 = {}, acc1 = {};
  for (int k = 0; k < H; k += 64) {
    bf16x16 a0 = load_frag(aRow + k + kA,      aRow + k + kA + 16);
    bf16x16 b0 = load_frag(bRow + k + kB,      bRow + k + kB + 16);
    bf16x16 a1 = load_frag(aRow + k + 32 + kA, aRow + k + 32 + kA + 16);
    bf16x16 b1 = load_frag(bRow + k + 32 + kB, bRow + k + 32 + kB + 16);
    acc0 = WMMA_BF16(a0, b0, acc0);
    acc1 = WMMA_BF16(a1, b1, acc1);
  }
#pragma unroll
  for (int i = 0; i < 8; ++i) acc0[i] += acc1[i];
  return acc0;
}

// ---------------- 1) weight transpose + fp32 -> bf16 ----------------
// dst[n*H + k] = src[k*H + n]   (so WMMA B fragments read contiguous K)
__global__ __launch_bounds__(256) void transpose_convert(
    const float* __restrict__ Wi, const float* __restrict__ Wh, const float* __restrict__ Wo,
    bf16* __restrict__ WiT, bf16* __restrict__ WhT, bf16* __restrict__ WoT) {
  __shared__ float tile[32][33];
  const float* src = (blockIdx.z == 0) ? Wi : (blockIdx.z == 1) ? Wh : Wo;
  bf16*        dst = (blockIdx.z == 0) ? WiT : (blockIdx.z == 1) ? WhT : WoT;

  int x = blockIdx.x * 32 + threadIdx.x;       // n
  int y = blockIdx.y * 32 + threadIdx.y;       // k (stepped by 8)
#pragma unroll
  for (int j = 0; j < 32; j += 8)
    tile[threadIdx.y + j][threadIdx.x] = src[(size_t)(y + j) * H + x];
  __syncthreads();
  int x2 = blockIdx.y * 32 + threadIdx.x;      // transposed col
  int y2 = blockIdx.x * 32 + threadIdx.y;      // transposed row
#pragma unroll
  for (int j = 0; j < 32; j += 8)
    dst[(size_t)(y2 + j) * H + x2] = f2bf(tile[threadIdx.x][threadIdx.y + j]);
}

// ---------------- 2) embedding gather -> bf16 X[t*B+b][h] ----------------
__global__ __launch_bounds__(256) void gather_embed(
    const int* __restrict__ input, const float* __restrict__ emb, bf16* __restrict__ X) {
  int blk = blockIdx.x;            // b*T + t   (input is [B,T])
  int b = blk / TSZ, t = blk % TSZ;
  int tok = input[blk];
  int idx = threadIdx.x * 4;       // 256 threads * 4 = 1024 = H
  const float4 v = *(const float4*)(emb + (size_t)tok * H + idx);
  bf16x4s o;
  o.v[0] = f2bf(v.x); o.v[1] = f2bf(v.y); o.v[2] = f2bf(v.z); o.v[3] = f2bf(v.w);
  *(bf16x4s*)(X + ((size_t)t * BSZ + b) * H + idx) = o;
}

// ---------------- 3) init hidden: fp32 -> bf16 ----------------
__global__ __launch_bounds__(256) void init_h(const float* __restrict__ h0f,
                                              bf16* __restrict__ h0b) {
  int i = blockIdx.x * 256 + threadIdx.x;
  h0b[i] = f2bf(h0f[i]);
}

// ---------------- 4) big GEMM: ix = X @ Wi + bi  (M=16384,K=N=1024) ------
// Each wave: 32x32 tile (2x2 of 16x16), block = 8 waves = 64x128 tile.
__global__ __launch_bounds__(256) void gemm_ix(
    const bf16* __restrict__ X, const bf16* __restrict__ WiT,
    const float* __restrict__ bi, float* __restrict__ ix) {
  const int wave = threadIdx.x >> 5;
  const int lane = threadIdx.x & 31;
  const int m0 = blockIdx.y * 64 + (wave >> 2) * 32;
  const int n0 = blockIdx.x * 128 + (wave & 3) * 32;
  const int kA = (lane >> 4) * 8;    // A K-chunk base per ISA A-layout
  const int kB = (lane >> 4) * 8;    // B K-chunk base per ISA B-layout
  const int mlane = lane & 15;

  const bf16* a0p = X + (size_t)(m0 + mlane) * H;
  const bf16* a1p = X + (size_t)(m0 + 16 + mlane) * H;
  const bf16* b0p = WiT + (size_t)(n0 + mlane) * H;
  const bf16* b1p = WiT + (size_t)(n0 + 16 + mlane) * H;

  f32x8 acc00 = {}, acc01 = {}, acc10 = {}, acc11 = {};
  for (int k = 0; k < H; k += 32) {
    bf16x16 a0 = load_frag(a0p + k + kA, a0p + k + kA + 16);
    bf16x16 a1 = load_frag(a1p + k + kA, a1p + k + kA + 16);
    bf16x16 b0 = load_frag(b0p + k + kB, b0p + k + kB + 16);
    bf16x16 b1 = load_frag(b1p + k + kB, b1p + k + kB + 16);
    acc00 = WMMA_BF16(a0, b0, acc00);
    acc01 = WMMA_BF16(a0, b1, acc01);
    acc10 = WMMA_BF16(a1, b0, acc10);
    acc11 = WMMA_BF16(a1, b1, acc11);
  }
  const int rsel = (lane >> 4) * 8;  // D layout: VGPR r -> M = r (+8 for hi lanes)
  const int nl = lane & 15;
#pragma unroll
  for (int r = 0; r < 8; ++r) {
    int mA = m0 + r + rsel, mB = m0 + 16 + r + rsel;
    int nA = n0 + nl,       nB = n0 + 16 + nl;
    ix[(size_t)mA * H + nA] = acc00[r] + bi[nA];
    ix[(size_t)mA * H + nB] = acc01[r] + bi[nB];
    ix[(size_t)mB * H + nA] = acc10[r] + bi[nA];
    ix[(size_t)mB * H + nB] = acc11[r] + bi[nB];
  }
}

// ---------------- 5) fused scan step ----------------
// Both halves depend only on hin = s(t-1), so they run concurrently:
//   waves/tiles   0..255 : s(t)    = tanh(ix_t + hin @ Wh + bh)  -> hout
//   waves/tiles 256..511 : out(t-1)=           hin @ Wo + bo     -> d_out
// 64 blocks x 8 waves = 512 tiles of 16x16 over [64 x 1024].
__global__ __launch_bounds__(256) void rnn_step_fused(
    const bf16* __restrict__ hin,
    const bf16* __restrict__ WhT, const float* __restrict__ ix_t,
    const float* __restrict__ bh, bf16* __restrict__ hout,
    float* __restrict__ hf32, int write_hf,
    const bf16* __restrict__ WoT, const float* __restrict__ bo,
    float* __restrict__ out, int tprev) {
  const int wave = threadIdx.x >> 5;
  const int lane = threadIdx.x & 31;
  const int tile = blockIdx.x * 8 + wave;
  const int kA = (lane >> 4) * 8;
  const int kB = (lane >> 4) * 8;
  const int mlane = lane & 15;
  const int rsel = (lane >> 4) * 8;
  const int nl = lane & 15;

  if (tile < 256) {
    // ---- hidden-state half: s(t) ----
    const int m0 = (tile >> 6) * 16;
    const int n0 = (tile & 63) * 16;
    const bf16* ap = hin + (size_t)(m0 + mlane) * H;
    const bf16* bp = WhT + (size_t)(n0 + mlane) * H;
    f32x8 acc = tile_gemm(ap, bp, kA, kB);
#pragma unroll
    for (int r = 0; r < 8; ++r) {
      int m = m0 + r + rsel;
      int n = n0 + nl;
      float v = acc[r] + ix_t[(size_t)m * H + n] + bh[n];
      float h = tanhf(v);
      hout[(size_t)m * H + n] = f2bf(h);
      if (write_hf) hf32[(size_t)m * H + n] = h;
    }
  } else {
    // ---- output half: out(t-1) = s(t-1) @ Wo + bo ----
    if (tprev < 0) return;                    // no output before step 0
    const int to = tile - 256;
    const int m0 = (to >> 6) * 16;
    const int n0 = (to & 63) * 16;
    const bf16* ap = hin + (size_t)(m0 + mlane) * H;
    const bf16* bp = WoT + (size_t)(n0 + mlane) * H;
    f32x8 acc = tile_gemm(ap, bp, kA, kB);
#pragma unroll
    for (int r = 0; r < 8; ++r) {
      int b = m0 + r + rsel;                  // batch row
      int n = n0 + nl;
      out[((size_t)b * TSZ + tprev) * OSZ + n] = acc[r] + bo[n];
    }
  }
}

// ---------------- 6) trailing output: out(T-1) = s(T-1) @ Wo + bo --------
__global__ __launch_bounds__(256) void rnn_step_out(
    const bf16* __restrict__ h, const bf16* __restrict__ WoT,
    const float* __restrict__ bo, float* __restrict__ out, int t) {
  const int wave = threadIdx.x >> 5;
  const int lane = threadIdx.x & 31;
  const int tile = blockIdx.x * 8 + wave;
  const int m0 = (tile >> 6) * 16;
  const int n0 = (tile & 63) * 16;
  const int kA = (lane >> 4) * 8;
  const int kB = (lane >> 4) * 8;
  const int mlane = lane & 15;

  const bf16* ap = h + (size_t)(m0 + mlane) * H;
  const bf16* bp = WoT + (size_t)(n0 + mlane) * H;
  f32x8 acc = tile_gemm(ap, bp, kA, kB);

  const int rsel = (lane >> 4) * 8;
  const int nl = lane & 15;
#pragma unroll
  for (int r = 0; r < 8; ++r) {
    int b = m0 + r + rsel;
    int n = n0 + nl;
    out[((size_t)b * TSZ + t) * OSZ + n] = acc[r] + bo[n];
  }
}

// ---------------- host-side orchestration ----------------
extern "C" void kernel_launch(void* const* d_in, const int* in_sizes, int n_in,
                              void* d_out, int out_size, void* d_ws, size_t ws_size,
                              hipStream_t stream) {
  const int*   input  = (const int*)  d_in[0];
  const float* hidden = (const float*)d_in[1];
  const float* emb    = (const float*)d_in[2];
  const float* Wi     = (const float*)d_in[3];
  const float* bi     = (const float*)d_in[4];
  const float* Wh     = (const float*)d_in[5];
  const float* bh     = (const float*)d_in[6];
  const float* Wo     = (const float*)d_in[7];
  const float* bo     = (const float*)d_in[8];
  float* out = (float*)d_out;

  char* ws = (char*)d_ws;
  size_t off = 0;
  bf16* WiT = (bf16*)(ws + off); off += (size_t)H * H * sizeof(bf16);
  bf16* WhT = (bf16*)(ws + off); off += (size_t)H * H * sizeof(bf16);
  bf16* WoT = (bf16*)(ws + off); off += (size_t)H * H * sizeof(bf16);
  bf16* X   = (bf16*)(ws + off); off += (size_t)TSZ * BSZ * H * sizeof(bf16);
  float* ix = (float*)(ws + off); off += (size_t)TSZ * BSZ * H * sizeof(float);
  bf16* hb0 = (bf16*)(ws + off); off += (size_t)BSZ * H * sizeof(bf16);
  bf16* hb1 = (bf16*)(ws + off); off += (size_t)BSZ * H * sizeof(bf16);
  float* hf = (float*)(ws + off); off += (size_t)BSZ * H * sizeof(float);
  bf16* hb[2] = { hb0, hb1 };

  // 1) weights -> bf16, transposed for contiguous WMMA B-fragment loads
  transpose_convert<<<dim3(H / 32, H / 32, 3), dim3(32, 8), 0, stream>>>(
      Wi, Wh, Wo, WiT, WhT, WoT);

  // 2) embedding gather to bf16 X[t*B+b][h]
  gather_embed<<<BSZ * TSZ, 256, 0, stream>>>(input, emb, X);

  // 3) initial hidden -> bf16 ping buffer 0
  init_h<<<(BSZ * H) / 256, 256, 0, stream>>>(hidden, hb0);

  // 4) ix = X @ Wi + bi (WMMA bf16, f32 accum)
  gemm_ix<<<dim3(H / 128, (TSZ * BSZ) / 64), 256, 0, stream>>>(X, WiT, bi, ix);

  // 5) fused sequential scan: one launch per step computes s(t) AND out(t-1)
  for (int t = 0; t < TSZ; ++t) {
    const bf16* hin  = hb[t & 1];
    bf16*       hout = hb[(t & 1) ^ 1];
    rnn_step_fused<<<64, 256, 0, stream>>>(
        hin, WhT, ix + (size_t)t * BSZ * H, bh, hout, hf, (t == TSZ - 1),
        WoT, bo, out, t - 1);
  }
  // trailing output for the last step: s(T-1) lives in hb[0] after t=255
  rnn_step_out<<<32, 256, 0, stream>>>(hb[(TSZ & 1) ^ (TSZ - 1) & 1 ? 0 : 0],
                                       WoT, bo, out, TSZ - 1);

  // 6) final hidden (fp32) appended after output_seq
  hipMemcpyAsync(out + (size_t)BSZ * TSZ * OSZ, hf,
                 (size_t)BSZ * H * sizeof(float),
                 hipMemcpyDeviceToDevice, stream);
}
